// MixtureOfBlocksAttention_76501957476957
// MI455X (gfx1250) — matrix operand order repaired and checked by reading, not compiled, not run on hardware
//
#include <hip/hip_runtime.h>

typedef __attribute__((ext_vector_type(16))) _Float16 v16h;
typedef __attribute__((ext_vector_type(8)))  float    v8f;

#define S_TOK  4096
#define NH     16
#define NKV    4
#define GRP    (NH / NKV)      // 4
#define HD     128
#define CHUNK  512
#define NCH    (S_TOK / CHUNK) // 8
#define QD     (NH * HD)       // 2048
#define KD     (NKV * HD)      // 512
#define NEGF   (-1.0e30f)
#define SCALEF (0.08838834764831845f)  // 1/sqrt(128)

// ---------------------------------------------------------------------------
// Kernel 1: per-chunk mean-pooled K  ->  km[nC][HKV][D]
// ---------------------------------------------------------------------------
__global__ __launch_bounds__(128)
void moba_kmean(const float* __restrict__ kmat, float* __restrict__ km)
{
  const int c  = blockIdx.x >> 2;   // chunk
  const int hk = blockIdx.x & 3;    // kv head
  const int d  = threadIdx.x;       // 0..127
  const float* p = kmat + (size_t)c * CHUNK * KD + hk * HD + d;
  float sum = 0.f;
  for (int i = 0; i < CHUNK; ++i) sum += p[(size_t)i * KD];
  km[(c * NKV + hk) * HD + d] = sum * (1.0f / CHUNK);
}

// ---------------------------------------------------------------------------
// Kernel 2: gates + top-3 selection -> sel[s*H+h] bitmask over 8 chunks
// One wave32 per (s,h); lanes split D into float4s.
// ---------------------------------------------------------------------------
__global__ __launch_bounds__(128)
void moba_gate(const float* __restrict__ q, const float* __restrict__ km,
               unsigned* __restrict__ sel)
{
  const int gtid = blockIdx.x * blockDim.x + threadIdx.x;
  const int wid  = gtid >> 5;
  const int lane = gtid & 31;
  if (wid >= S_TOK * NH) return;
  const int s  = wid / NH;
  const int h  = wid % NH;
  const int hk = h / GRP;
  const int cq = s / CHUNK;

  const float4 qv = *(const float4*)(q + (size_t)s * QD + h * HD + lane * 4);

  float g[NCH];
#pragma unroll
  for (int c = 0; c < NCH; ++c) {
    const float4 kv = *(const float4*)(km + (c * NKV + hk) * HD + lane * 4);
    float p = qv.x * kv.x + qv.y * kv.y + qv.z * kv.z + qv.w * kv.w;
#pragma unroll
    for (int off = 16; off; off >>= 1) p += __shfl_xor(p, off, 32);
    g[c] = (c < cq) ? p * SCALEF : NEGF;
  }
  // top-3 threshold (m3 = 3rd largest among the 8 masked gates)
  float m1 = -3.4e38f, m2 = -3.4e38f, m3 = -3.4e38f;
#pragma unroll
  for (int c = 0; c < NCH; ++c) {
    const float x = g[c];
    if (x > m1)      { m3 = m2; m2 = m1; m1 = x; }
    else if (x > m2) { m3 = m2; m2 = x; }
    else if (x > m3) { m3 = x; }
  }
  unsigned mask = 0;
#pragma unroll
  for (int c = 0; c < NCH; ++c)
    if ((c < cq) && (g[c] >= m3) && (g[c] > 0.5f * NEGF)) mask |= (1u << c);
  if (lane == 0) sel[wid] = mask;
}

// ---------------------------------------------------------------------------
// Kernel 3: block-sparse flash attention. One wave per (16-query tile, head).
// QK^T and P*V via v_wmma_f32_16x16x32_f16, f32 online-softmax accumulators.
// ---------------------------------------------------------------------------
__global__ __launch_bounds__(32)
void moba_attn(const float* __restrict__ q, const float* __restrict__ kmat,
               const float* __restrict__ vmat, const unsigned* __restrict__ sel,
               float* __restrict__ out)
{
  const int lane = threadIdx.x;     // 0..31
  const int qb   = blockIdx.x * 16; // first query row of tile
  const int h    = blockIdx.y;
  const int hk   = h / GRP;
  const int n    = lane & 15;       // column / row-in-A index
  const int lh   = lane >> 4;       // lane half

  __shared__ _Float16 Qs[16 * 132];
  __shared__ _Float16 Ks[32 * 132];
  __shared__ _Float16 Vs[128 * 34];   // transposed: [d][key]
  __shared__ _Float16 Ps[16 * 34];
  __shared__ unsigned SelS[16];

  // ---- stage Q tile (16 x 128) as f16 ----
  for (int idx = lane; idx < (16 * 128) / 4; idx += 32) {
    const int i = idx * 4, m = i >> 7, d = i & 127;
    const float4 f = *(const float4*)(q + (size_t)(qb + m) * QD + h * HD + d);
    Qs[m * 132 + d + 0] = (_Float16)f.x;
    Qs[m * 132 + d + 1] = (_Float16)f.y;
    Qs[m * 132 + d + 2] = (_Float16)f.z;
    Qs[m * 132 + d + 3] = (_Float16)f.w;
  }
  if (lane < 16) SelS[lane] = sel[(size_t)(qb + lane) * NH + h];
  __syncthreads();

  // ---- Q A-fragments, one per 32-wide K-dim step ----
  v16h aQ[4];
#pragma unroll
  for (int t = 0; t < 4; ++t)
#pragma unroll
    for (int vv = 0; vv < 8; ++vv) {
      const int k0 = t * 32 + lh * 16 + 2 * vv;
      aQ[t][2 * vv]     = Qs[n * 132 + k0];
      aQ[t][2 * vv + 1] = Qs[n * 132 + k0 + 1];
    }

  unsigned rowsel[8];
  unsigned unionSel = 0;
#pragma unroll
  for (int r = 0; r < 8; ++r) rowsel[r] = SelS[r + 8 * lh];
  for (int i = 0; i < 16; ++i) unionSel |= SelS[i];

  float mr[8], lr[8], alph[8];
  v8f   O[8];
#pragma unroll
  for (int r = 0; r < 8; ++r) { mr[r] = NEGF; lr[r] = 0.f; }
#pragma unroll
  for (int d = 0; d < 8; ++d)
#pragma unroll
    for (int r = 0; r < 8; ++r) O[d][r] = 0.f;

  const int cq   = qb / CHUNK;
  const int qoff = qb - cq * CHUNK;

  for (int c = 0; c <= cq; ++c) {
    const bool iscur = (c == cq);
    if (!iscur && !((unionSel >> c) & 1u)) continue;  // wave-uniform skip
    const int nkb = iscur ? (qoff / 32 + 1) : (CHUNK / 32);

    for (int kb = 0; kb < nkb; ++kb) {
      const int keybase = c * CHUNK + kb * 32;
      __syncthreads();  // previous block's LDS reads complete before restage

      // ---- stage K (32x128) and V^T (128x32) as f16 ----
      for (int idx = lane; idx < (32 * 128) / 4; idx += 32) {
        const int i = idx * 4, row = i >> 7, d = i & 127;
        const size_t gofs = (size_t)(keybase + row) * KD + hk * HD + d;
        const float4 fk = *(const float4*)(kmat + gofs);
        Ks[row * 132 + d + 0] = (_Float16)fk.x;
        Ks[row * 132 + d + 1] = (_Float16)fk.y;
        Ks[row * 132 + d + 2] = (_Float16)fk.z;
        Ks[row * 132 + d + 3] = (_Float16)fk.w;
        const float4 fv = *(const float4*)(vmat + gofs);
        Vs[(d + 0) * 34 + row] = (_Float16)fv.x;
        Vs[(d + 1) * 34 + row] = (_Float16)fv.y;
        Vs[(d + 2) * 34 + row] = (_Float16)fv.z;
        Vs[(d + 3) * 34 + row] = (_Float16)fv.w;
      }
      __syncthreads();

      // ---- S block (16 queries x 32 keys), two 16x16 C-tiles ----
      v8f Sa0 = {}; v8f Sa1 = {};
#pragma unroll
      for (int t = 0; t < 4; ++t) {
        v16h b0, b1;
#pragma unroll
        for (int vv = 0; vv < 8; ++vv) {
          const int d0 = t * 32 + lh * 16 + 2 * vv;  // K index within D
          b0[2 * vv]     = Ks[n * 132 + d0];          // key row = n
          b0[2 * vv + 1] = Ks[n * 132 + d0 + 1];
          b1[2 * vv]     = Ks[(16 + n) * 132 + d0];   // key row = 16+n
          b1[2 * vv + 1] = Ks[(16 + n) * 132 + d0 + 1];
        }
        Sa0 = __builtin_amdgcn_wmma_f32_16x16x32_f16(false, aQ[t], false, b0,
                                                     (short)0, Sa0, false, false);
        Sa1 = __builtin_amdgcn_wmma_f32_16x16x32_f16(false, aQ[t], false, b1,
                                                     (short)0, Sa1, false, false);
      }

      // ---- online softmax over this 32-key block ----
#pragma unroll
      for (int r = 0; r < 8; ++r) {
        const int  mrow = r + 8 * lh;
        const int  qpos = qb + mrow;
        const bool rsel = iscur ? true : (((rowsel[r] >> c) & 1u) != 0u);
        float s0 = Sa0[r] * SCALEF;
        float s1 = Sa1[r] * SCALEF;
        const bool a0 = rsel && (!iscur || (keybase + n)      <= qpos);
        const bool a1 = rsel && (!iscur || (keybase + 16 + n) <= qpos);
        s0 = a0 ? s0 : NEGF;
        s1 = a1 ? s1 : NEGF;
        float tm = fmaxf(s0, s1);
#pragma unroll
        for (int off = 8; off; off >>= 1) tm = fmaxf(tm, __shfl_xor(tm, off, 32));
        const float mn = fmaxf(mr[r], tm);
        const float al = __expf(mr[r] - mn);
        const float p0 = a0 ? __expf(s0 - mn) : 0.f;
        const float p1 = a1 ? __expf(s1 - mn) : 0.f;
        float ps = p0 + p1;
#pragma unroll
        for (int off = 8; off; off >>= 1) ps += __shfl_xor(ps, off, 32);
        lr[r] = lr[r] * al + ps;
        mr[r] = mn;
        alph[r] = al;
        Ps[mrow * 34 + n]      = (_Float16)p0;
        Ps[mrow * 34 + 16 + n] = (_Float16)p1;
      }
#pragma unroll
      for (int d = 0; d < 8; ++d)
#pragma unroll
        for (int r = 0; r < 8; ++r) O[d][r] *= alph[r];
      __syncthreads();  // Ps fully written before A-fragment gather

      // ---- P (16x32) A-fragment, then P*V into O ----
      v16h aP;
#pragma unroll
      for (int vv = 0; vv < 8; ++vv) {
        const int k0 = lh * 16 + 2 * vv;
        aP[2 * vv]     = Ps[n * 34 + k0];
        aP[2 * vv + 1] = Ps[n * 34 + k0 + 1];
      }
#pragma unroll
      for (int ds = 0; ds < 8; ++ds) {
        v16h bV;
#pragma unroll
        for (int vv = 0; vv < 8; ++vv) {
          const int k0 = lh * 16 + 2 * vv;            // key index 0..31
          bV[2 * vv]     = Vs[(ds * 16 + n) * 34 + k0];
          bV[2 * vv + 1] = Vs[(ds * 16 + n) * 34 + k0 + 1];
        }
        O[ds] = __builtin_amdgcn_wmma_f32_16x16x32_f16(false, aP, false, bV,
                                                       (short)0, O[ds], false, false);
      }
    }
  }

  // ---- normalize and store ----
#pragma unroll
  for (int ds = 0; ds < 8; ++ds)
#pragma unroll
    for (int r = 0; r < 8; ++r) {
      const int mrow = r + 8 * lh;
      out[(size_t)(qb + mrow) * QD + h * HD + ds * 16 + n] = O[ds][r] / lr[r];
    }
}

// ---------------------------------------------------------------------------
extern "C" void kernel_launch(void* const* d_in, const int* in_sizes, int n_in,
                              void* d_out, int out_size, void* d_ws, size_t ws_size,
                              hipStream_t stream)
{
  (void)in_sizes; (void)n_in; (void)out_size; (void)ws_size;
  const float* q = (const float*)d_in[0];
  const float* k = (const float*)d_in[1];
  const float* v = (const float*)d_in[2];
  float* out = (float*)d_out;

  float*    km  = (float*)d_ws;                                      // 16 KB
  unsigned* sel = (unsigned*)((char*)d_ws +
                              (size_t)NCH * NKV * HD * sizeof(float)); // 256 KB

  moba_kmean<<<NCH * NKV, 128, 0, stream>>>(k, km);
  moba_gate<<<(S_TOK * NH * 32) / 128, 128, 0, stream>>>(q, km, sel);
  moba_attn<<<dim3(S_TOK / 16, NH), 32, 0, stream>>>(q, k, v, sel, out);
}
